// GenreAwareRefinement_36223754175137
// MI455X (gfx1250) — matrix operand order repaired and checked by reading, not compiled, not run on hardware
//
#include <hip/hip_runtime.h>
#include <hip/hip_bf16.h>

typedef __bf16 bf16_t;
typedef __attribute__((ext_vector_type(16))) __bf16 v16bf;
typedef __attribute__((ext_vector_type(8)))  __bf16 v8bf;
typedef __attribute__((ext_vector_type(8)))  float  v8f;
typedef int v4i_t __attribute__((vector_size(16)));   // matches async builtin param

#define B_ 16384
#define D_ 512
#define H_ 128
#define G_ 18

#define AS1 __attribute__((address_space(1)))
#define AS3 __attribute__((address_space(3)))

// ---------- CDNA5 async global->LDS copy (ASYNCcnt path), with fallback ----------

#if defined(__gfx1250__) && __has_builtin(__builtin_amdgcn_global_load_async_to_lds_b128)
#define HAS_ASYNC_LDS 1
#else
#define HAS_ASYNC_LDS 0
#endif

__device__ __forceinline__ void async_copy16(const bf16_t* g, bf16_t* l) {
#if HAS_ASYNC_LDS
  __builtin_amdgcn_global_load_async_to_lds_b128(
      (AS1 v4i_t*)g, (AS3 v4i_t*)l, 0, 0);
#else
  *(uint4*)l = *(const uint4*)g;   // vmem b128 load + ds b128 store
#endif
}

__device__ __forceinline__ void async_wait() {
#if HAS_ASYNC_LDS
#if __has_builtin(__builtin_amdgcn_s_wait_asynccnt)
  __builtin_amdgcn_s_wait_asynccnt(0);
#else
  asm volatile("s_wait_asynccnt 0" ::: "memory");
#endif
#endif
}

// ---------- helpers ----------

__device__ __forceinline__ bf16_t f2bf(float f) {
  union { float f; unsigned int u; } v; v.f = f;
  unsigned int r = v.u + 0x7FFFu + ((v.u >> 16) & 1u);   // round-to-nearest-even
  unsigned short h = (unsigned short)(r >> 16);
  return __builtin_bit_cast(bf16_t, h);
}

__device__ __forceinline__ v8f v8f_zero() {
  v8f z = {0.f,0.f,0.f,0.f,0.f,0.f,0.f,0.f};
  return z;
}

// A-matrix 16x32 bf16 fragment from LDS (row-major, `stride` halfs per row).
// ISA 7.12.2: lanes 0-15 hold K=[0..7],[16..23]; lanes 16-31 hold K=[8..15],[24..31].
__device__ __forceinline__ v16bf load_a_frag(const bf16_t* base, int stride,
                                             int m0, int k0, int lane) {
  int mm = m0 + (lane & 15);
  int kk = k0 + ((lane >> 4) << 3);
  const bf16_t* p = base + mm * stride + kk;
  v8bf lo = *(const v8bf*)(p);
  v8bf hi = *(const v8bf*)(p + 16);
  return __builtin_shufflevector(lo, hi, 0,1,2,3,4,5,6,7,8,9,10,11,12,13,14,15);
}

// B-matrix 32x16 bf16 fragment from LDS panel stored [n][k] (32 halfs per n).
__device__ __forceinline__ v16bf load_b_frag(const bf16_t* wlds, int n0, int lane) {
  int nn = n0 + (lane & 15);
  int kk = (lane >> 4) << 3;
  const bf16_t* q = wlds + nn * 32 + kk;
  v8bf lo = *(const v8bf*)(q);
  v8bf hi = *(const v8bf*)(q + 16);
  return __builtin_shufflevector(lo, hi, 0,1,2,3,4,5,6,7,8,9,10,11,12,13,14,15);
}

__device__ __forceinline__ v8f wmma_bf16(v16bf a, v16bf b, v8f c) {
  return __builtin_amdgcn_wmma_f32_16x16x32_bf16(false, a, false, b,
                                                 (short)0, c, false, false);
}

// ---------- kernel W: one-time weight repack fp32 -> bf16, panel-blocked ----------
// src: [g][K][N] fp32 row-major. dst: [g][K/32][N][32] bf16 (contiguous panels so
// k-step staging is a pure async bulk copy).

__global__ __launch_bounds__(256) void repack_kernel(
    const float* __restrict__ src, bf16_t* __restrict__ dst,
    int K, int N, int total)
{
  int idx = blockIdx.x * 256 + threadIdx.x;
  if (idx >= total) return;
  int n = idx % N;
  int k = (idx / N) % K;
  int g = idx / (N * K);
  int p = k >> 5, kk = k & 31;
  dst[(size_t)g * K * N + (size_t)p * N * 32 + n * 32 + kk] = f2bf(src[idx]);
}

// ---------- kernel A: attention softmax * genre mask, plus fp32->bf16 X ----------

__global__ __launch_bounds__(256) void attn_kernel(
    const float* __restrict__ X, const float* __restrict__ GV,
    const float* __restrict__ Wa, const float* __restrict__ ba,
    float* __restrict__ wmask, bf16_t* __restrict__ Xb)
{
  const int wave = threadIdx.x >> 5;
  const int lane = threadIdx.x & 31;
  const int row  = blockIdx.x * 8 + wave;
  const float* xr = X + (size_t)row * D_;

  float xv[16];
#pragma unroll
  for (int i = 0; i < 16; ++i) {
    int d = lane + 32 * i;
    xv[i] = xr[d];
    Xb[(size_t)row * D_ + d] = f2bf(xv[i]);
  }

  float logit[G_];
#pragma unroll 1
  for (int g = 0; g < G_; ++g) {
    float s = 0.f;
#pragma unroll
    for (int i = 0; i < 16; ++i)
      s += xv[i] * Wa[(size_t)(lane + 32 * i) * G_ + g];
#pragma unroll
    for (int off = 16; off > 0; off >>= 1)
      s += __shfl_xor(s, off, 32);
    logit[g] = s + ba[g];
  }

  float mx = logit[0];
#pragma unroll
  for (int g = 1; g < G_; ++g) mx = fmaxf(mx, logit[g]);
  float den = 0.f;
#pragma unroll
  for (int g = 0; g < G_; ++g) { logit[g] = __expf(logit[g] - mx); den += logit[g]; }
  float inv = 1.f / den;

  if (lane < G_) {
    float gv = GV[(size_t)row * G_ + lane];
    wmask[(size_t)row * G_ + lane] = (gv > 0.f) ? logit[lane] * inv * gv : 0.f;
  }
}

// ---------- kernel B: fused 3-layer experts, async double-buffered weight panels ----
// block = 256 threads = 8 waves tiled 2(M) x 4(N); M_TILE = 64 rows.
// LDS: X 64x512 | h1 64x256 | h2 64x128 | Wbuf 2 x (256x32) | wrow 64 f32  (~147 KB)

__global__ __launch_bounds__(256) void expert_kernel(
    const bf16_t* __restrict__ Xb, const float* __restrict__ wmask,
    const bf16_t* __restrict__ Wb1, const float* __restrict__ b1,
    const bf16_t* __restrict__ Wb2, const float* __restrict__ b2,
    const bf16_t* __restrict__ Wb3, const float* __restrict__ b3,
    bf16_t* __restrict__ refb)
{
  extern __shared__ char smem[];
  bf16_t* Xlds  = (bf16_t*)smem;              // 64*512
  bf16_t* h1lds = Xlds  + 64 * 512;           // 64*256
  bf16_t* h2lds = h1lds + 64 * 256;           // 64*128
  bf16_t* Wbuf0 = h2lds + 64 * 128;           // 256*32
  bf16_t* Wbuf1 = Wbuf0 + 256 * 32;           // 256*32
  float*  wrow  = (float*)(Wbuf1 + 256 * 32); // 64

  const int tid  = threadIdx.x;
  const int lane = tid & 31;
  const int wave = tid >> 5;
  const int mg   = wave >> 2;   // 0..1 : row group of 32
  const int ng   = wave & 3;    // 0..3 : col group
  const int row0 = blockIdx.x * 64;

  bf16_t* wb[2] = {Wbuf0, Wbuf1};

  // async-copy the bf16 X tile (contiguous 64 KB)
  {
    const bf16_t* src = Xb + (size_t)row0 * D_;
#pragma unroll
    for (int i = 0; i < 16; ++i) {
      int off = (tid + 256 * i) * 8;
      async_copy16(src + off, Xlds + off);
    }
  }

  v8f refacc[2][2];
#pragma unroll
  for (int a = 0; a < 2; ++a)
#pragma unroll
    for (int c = 0; c < 2; ++c) refacc[a][c] = v8f_zero();

  for (int g = 0; g < G_; ++g) {
    const bf16_t* W1g = Wb1 + (size_t)g * 512 * 256;  // 16 panels of 256x32
    const bf16_t* W2g = Wb2 + (size_t)g * 256 * 128;  // 8 panels of 128x32
    const bf16_t* W3g = Wb3 + (size_t)g * 128 * 128;  // 4 panels of 128x32

    __syncthreads();
    if (tid < 64) wrow[tid] = wmask[(size_t)(row0 + tid) * G_ + g];
    if (g + 1 < G_)
      __builtin_prefetch(Wb1 + (size_t)(g + 1) * 512 * 256 + tid * 8, 0, 1);

    // ===== GEMM1: h1 = relu(X[64,512] @ W1g[512,256] + b1), 16 k-panels =====
    v8f acc1[2][4];
#pragma unroll
    for (int mt = 0; mt < 2; ++mt)
#pragma unroll
      for (int ct = 0; ct < 4; ++ct) acc1[mt][ct] = v8f_zero();

#pragma unroll
    for (int i = 0; i < 4; ++i) {
      int off = (tid + 256 * i) * 8;
      async_copy16(W1g + off, wb[0] + off);
    }
    for (int p = 0; p < 16; ++p) {
      async_wait();
      __syncthreads();
      if (p + 1 < 16) {
        const bf16_t* srcp = W1g + (size_t)(p + 1) * 256 * 32;
        bf16_t* dstp = wb[(p + 1) & 1];
#pragma unroll
        for (int i = 0; i < 4; ++i) {
          int off = (tid + 256 * i) * 8;
          async_copy16(srcp + off, dstp + off);
        }
      }
      const bf16_t* wp = wb[p & 1];
      int k0 = p * 32;
#pragma unroll
      for (int mt = 0; mt < 2; ++mt) {
        v16bf a = load_a_frag(Xlds, 512, mg * 32 + mt * 16, k0, lane);
#pragma unroll
        for (int ct = 0; ct < 4; ++ct) {
          v16bf b = load_b_frag(wp, ng * 64 + ct * 16, lane);
          acc1[mt][ct] = wmma_bf16(a, b, acc1[mt][ct]);
        }
      }
    }
    __syncthreads();
#pragma unroll
    for (int mt = 0; mt < 2; ++mt)
#pragma unroll
      for (int ct = 0; ct < 4; ++ct) {
        int col   = ng * 64 + ct * 16 + (lane & 15);
        float bia = b1[(size_t)g * 256 + col];
        int rbase = mg * 32 + mt * 16 + ((lane >> 4) << 3);
#pragma unroll
        for (int j = 0; j < 8; ++j) {
          float v = acc1[mt][ct][j] + bia;
          h1lds[(rbase + j) * 256 + col] = f2bf(v > 0.f ? v : 0.f);
        }
      }

    // ===== GEMM2: h2 = relu(h1[64,256] @ W2g[256,128] + b2), 8 k-panels =====
    v8f acc2[2][2];
#pragma unroll
    for (int mt = 0; mt < 2; ++mt)
#pragma unroll
      for (int ct = 0; ct < 2; ++ct) acc2[mt][ct] = v8f_zero();

    __syncthreads();
#pragma unroll
    for (int i = 0; i < 2; ++i) {
      int off = (tid + 256 * i) * 8;
      async_copy16(W2g + off, wb[0] + off);
    }
    for (int p = 0; p < 8; ++p) {
      async_wait();
      __syncthreads();
      if (p + 1 < 8) {
        const bf16_t* srcp = W2g + (size_t)(p + 1) * 128 * 32;
        bf16_t* dstp = wb[(p + 1) & 1];
#pragma unroll
        for (int i = 0; i < 2; ++i) {
          int off = (tid + 256 * i) * 8;
          async_copy16(srcp + off, dstp + off);
        }
      }
      const bf16_t* wp = wb[p & 1];
      int k0 = p * 32;
#pragma unroll
      for (int mt = 0; mt < 2; ++mt) {
        v16bf a = load_a_frag(h1lds, 256, mg * 32 + mt * 16, k0, lane);
#pragma unroll
        for (int ct = 0; ct < 2; ++ct) {
          v16bf b = load_b_frag(wp, ng * 32 + ct * 16, lane);
          acc2[mt][ct] = wmma_bf16(a, b, acc2[mt][ct]);
        }
      }
    }
    __syncthreads();
#pragma unroll
    for (int mt = 0; mt < 2; ++mt)
#pragma unroll
      for (int ct = 0; ct < 2; ++ct) {
        int col   = ng * 32 + ct * 16 + (lane & 15);
        float bia = b2[(size_t)g * 128 + col];
        int rbase = mg * 32 + mt * 16 + ((lane >> 4) << 3);
#pragma unroll
        for (int j = 0; j < 8; ++j) {
          float v = acc2[mt][ct][j] + bia;
          h2lds[(rbase + j) * 128 + col] = f2bf(v > 0.f ? v : 0.f);
        }
      }

    // ===== GEMM3: h3 = h2[64,128] @ W3g[128,128] + b3; refacc += w[row]*h3 =====
    v8f acc3[2][2];
#pragma unroll
    for (int mt = 0; mt < 2; ++mt)
#pragma unroll
      for (int ct = 0; ct < 2; ++ct) acc3[mt][ct] = v8f_zero();

    __syncthreads();
#pragma unroll
    for (int i = 0; i < 2; ++i) {
      int off = (tid + 256 * i) * 8;
      async_copy16(W3g + off, wb[0] + off);
    }
    for (int p = 0; p < 4; ++p) {
      async_wait();
      __syncthreads();
      if (p + 1 < 4) {
        const bf16_t* srcp = W3g + (size_t)(p + 1) * 128 * 32;
        bf16_t* dstp = wb[(p + 1) & 1];
#pragma unroll
        for (int i = 0; i < 2; ++i) {
          int off = (tid + 256 * i) * 8;
          async_copy16(srcp + off, dstp + off);
        }
      }
      const bf16_t* wp = wb[p & 1];
      int k0 = p * 32;
#pragma unroll
      for (int mt = 0; mt < 2; ++mt) {
        v16bf a = load_a_frag(h2lds, 128, mg * 32 + mt * 16, k0, lane);
#pragma unroll
        for (int ct = 0; ct < 2; ++ct) {
          v16bf b = load_b_frag(wp, ng * 32 + ct * 16, lane);
          acc3[mt][ct] = wmma_bf16(a, b, acc3[mt][ct]);
        }
      }
    }
#pragma unroll
    for (int mt = 0; mt < 2; ++mt)
#pragma unroll
      for (int ct = 0; ct < 2; ++ct) {
        int col   = ng * 32 + ct * 16 + (lane & 15);
        float bia = b3[(size_t)g * 128 + col];
        int rbase = mg * 32 + mt * 16 + ((lane >> 4) << 3);
#pragma unroll
        for (int j = 0; j < 8; ++j) {
          float h3v = acc3[mt][ct][j] + bia;
          refacc[mt][ct][j] += wrow[rbase + j] * h3v;
        }
      }
  } // genre loop

#pragma unroll
  for (int mt = 0; mt < 2; ++mt)
#pragma unroll
    for (int ct = 0; ct < 2; ++ct) {
      int col   = ng * 32 + ct * 16 + (lane & 15);
      int rbase = mg * 32 + mt * 16 + ((lane >> 4) << 3);
#pragma unroll
      for (int j = 0; j < 8; ++j)
        refb[(size_t)(row0 + rbase + j) * H_ + col] = f2bf(refacc[mt][ct][j]);
    }
}

// ---------- kernel C: out = relu([X | ref] @ Wagg + bagg), K = 640 ----------
// block = 256 threads = 8 waves; M_TILE = 32; wave owns a 64-col strip; async
// double-buffered A (32x32) and B (512x32) panels.  LDS ~68 KB.

__global__ __launch_bounds__(256) void agg_kernel(
    const bf16_t* __restrict__ Xb, const bf16_t* __restrict__ refb,
    const bf16_t* __restrict__ Wba, const float* __restrict__ bagg,
    float* __restrict__ out)
{
  extern __shared__ char smem[];
  bf16_t* Blds = (bf16_t*)smem;        // 2 x 512*32
  bf16_t* Alds = Blds + 2 * 512 * 32;  // 2 x 32*32

  const int tid  = threadIdx.x;
  const int lane = tid & 31;
  const int wave = tid >> 5;
  const int row0 = blockIdx.x * 32;

  bf16_t* bb[2] = {Blds, Blds + 512 * 32};
  bf16_t* ab[2] = {Alds, Alds + 32 * 32};

  v8f acc[2][4];
#pragma unroll
  for (int mt = 0; mt < 2; ++mt)
#pragma unroll
    for (int ct = 0; ct < 4; ++ct) acc[mt][ct] = v8f_zero();

  auto stage = [&](int p, int bufsel) {
    const bf16_t* srcB = Wba + (size_t)p * 512 * 32;
    bf16_t* dstB = bb[bufsel];
#pragma unroll
    for (int i = 0; i < 8; ++i) {
      int off = (tid + 256 * i) * 8;
      async_copy16(srcB + off, dstB + off);
    }
    if (tid < 128) {
      int r = tid >> 2, part = tid & 3;
      int k0 = p * 32;
      const bf16_t* srcA = (k0 < 512)
          ? Xb   + (size_t)(row0 + r) * D_ + k0 + part * 8
          : refb + (size_t)(row0 + r) * H_ + (k0 - 512) + part * 8;
      async_copy16(srcA, ab[bufsel] + tid * 8);
    }
  };

  stage(0, 0);
  for (int p = 0; p < 20; ++p) {
    async_wait();
    __syncthreads();
    if (p + 1 < 20) stage(p + 1, (p + 1) & 1);
    const bf16_t* bp = bb[p & 1];
    const bf16_t* ap = ab[p & 1];
#pragma unroll
    for (int mt = 0; mt < 2; ++mt) {
      v16bf a = load_a_frag(ap, 32, mt * 16, 0, lane);
#pragma unroll
      for (int ct = 0; ct < 4; ++ct) {
        v16bf b = load_b_frag(bp, wave * 64 + ct * 16, lane);
        acc[mt][ct] = wmma_bf16(a, b, acc[mt][ct]);
      }
    }
  }

#pragma unroll
  for (int mt = 0; mt < 2; ++mt)
#pragma unroll
    for (int ct = 0; ct < 4; ++ct) {
      int col   = wave * 64 + ct * 16 + (lane & 15);
      float bia = bagg[col];
      int rbase = mt * 16 + ((lane >> 4) << 3);
#pragma unroll
      for (int j = 0; j < 8; ++j) {
        float v = acc[mt][ct][j] + bia;
        out[(size_t)(row0 + rbase + j) * 512 + col] = v > 0.f ? v : 0.f;
      }
    }
}

// ---------- launch ----------

extern "C" void kernel_launch(void* const* d_in, const int* in_sizes, int n_in,
                              void* d_out, int out_size, void* d_ws, size_t ws_size,
                              hipStream_t stream) {
  (void)in_sizes; (void)n_in; (void)out_size; (void)ws_size;
  const float* X    = (const float*)d_in[0];
  const float* GV   = (const float*)d_in[1];
  const float* W1   = (const float*)d_in[2];
  const float* b1   = (const float*)d_in[3];
  const float* W2   = (const float*)d_in[4];
  const float* b2   = (const float*)d_in[5];
  const float* W3   = (const float*)d_in[6];
  const float* b3   = (const float*)d_in[7];
  const float* Wa   = (const float*)d_in[8];
  const float* ba   = (const float*)d_in[9];
  const float* Wagg = (const float*)d_in[10];
  const float* bagg = (const float*)d_in[11];
  float* out = (float*)d_out;

  // workspace layout
  float*  wmask = (float*)d_ws;                        // B*G f32
  bf16_t* Xb    = (bf16_t*)(wmask + (size_t)B_ * G_);  // B*D
  bf16_t* refb  = Xb + (size_t)B_ * D_;                // B*H
  bf16_t* Wb1   = refb + (size_t)B_ * H_;              // G*512*256
  bf16_t* Wb2   = Wb1 + (size_t)G_ * 512 * 256;        // G*256*128
  bf16_t* Wb3   = Wb2 + (size_t)G_ * 256 * 128;        // G*128*128
  bf16_t* Wba   = Wb3 + (size_t)G_ * 128 * 128;        // 640*512

  // one-time weight repack (bf16, panel-blocked [K/32][N][32])
  int t1 = G_ * 512 * 256, t2 = G_ * 256 * 128, t3 = G_ * 128 * 128, t4 = 640 * 512;
  repack_kernel<<<(t1 + 255) / 256, 256, 0, stream>>>(W1, Wb1, 512, 256, t1);
  repack_kernel<<<(t2 + 255) / 256, 256, 0, stream>>>(W2, Wb2, 256, 128, t2);
  repack_kernel<<<(t3 + 255) / 256, 256, 0, stream>>>(W3, Wb3, 128, 128, t3);
  repack_kernel<<<(t4 + 255) / 256, 256, 0, stream>>>(Wagg, Wba, 640, 512, t4);

  attn_kernel<<<B_ / 8, 256, 0, stream>>>(X, GV, Wa, ba, wmask, Xb);

  size_t lds_b = (size_t)(64 * 512 + 64 * 256 + 64 * 128 + 2 * 256 * 32) * sizeof(bf16_t)
               + 64 * sizeof(float);
  expert_kernel<<<B_ / 64, 256, lds_b, stream>>>(Xb, wmask, Wb1, b1, Wb2, b2, Wb3, b3, refb);

  size_t lds_c = (size_t)(2 * 512 * 32 + 2 * 32 * 32) * sizeof(bf16_t);
  agg_kernel<<<B_ / 32, 256, lds_c, stream>>>(Xb, refb, Wba, bagg, out);
}
